// ComplementConstraint_62878321214076
// MI455X (gfx1250) — matrix operand order repaired and checked by reading, not compiled, not run on hardware
//
#include <hip/hip_runtime.h>
#include <math.h>

typedef __attribute__((ext_vector_type(4)))  float  f4;
typedef __attribute__((ext_vector_type(8)))  float  v8f;
typedef __attribute__((ext_vector_type(4)))  __bf16 v4bf;
typedef __attribute__((ext_vector_type(8)))  __bf16 v8bf;
typedef __attribute__((ext_vector_type(16))) __bf16 v16bf;

#define BATCH 8192
#define DK    128
#define NC    10000
#define TM    64
#define TN    128
#define TKC   64                       // K chunk staged in LDS per iteration
#define APITCH 72                      // bf16 row pitch: 144 B -> 16B aligned, bank-spread
#define NTILES ((NC + TN - 1) / TN)    // 79
#define WT_ROWS (NTILES * TN)          // 10112 (padded, zero-filled)

// ---------------------------------------------------------------------------
// Kernel 0a: split x (f32) -> xhi, xlo (bf16), same [BATCH][DK] layout.
// ---------------------------------------------------------------------------
__global__ __launch_bounds__(256)
void split_x_kernel(const float* __restrict__ x, __bf16* __restrict__ xhi,
                    __bf16* __restrict__ xlo)
{
    const int nq = BATCH * DK / 4;
    for (int q = blockIdx.x * 256 + threadIdx.x; q < nq; q += gridDim.x * 256) {
        f4 v = ((const f4*)x)[q];
        v4bf h, l;
        h[0] = (__bf16)v.x; l[0] = (__bf16)(v.x - (float)h[0]);
        h[1] = (__bf16)v.y; l[1] = (__bf16)(v.y - (float)h[1]);
        h[2] = (__bf16)v.z; l[2] = (__bf16)(v.z - (float)h[2]);
        h[3] = (__bf16)v.w; l[3] = (__bf16)(v.w - (float)h[3]);
        ((v4bf*)xhi)[q] = h;
        ((v4bf*)xlo)[q] = l;
    }
}

// ---------------------------------------------------------------------------
// Kernel 0b: transpose + split W (f32, [DK][NC]) -> WhiT, WloT (bf16, [WT_ROWS][DK]),
// rows >= NC zero-filled so the GEMM B-staging needs no bounds checks.
// ---------------------------------------------------------------------------
__global__ __launch_bounds__(256)
void transpose_split_w(const float* __restrict__ W, __bf16* __restrict__ WhiT,
                       __bf16* __restrict__ WloT)
{
    __shared__ __bf16 th[32][DK + 8];   // pitch 136 -> 272 B rows, 16B aligned
    __shared__ __bf16 tl[32][DK + 8];
    const int tid = threadIdx.x;
    const int n0  = blockIdx.x * 32;

    // read phase: coalesced along n
    for (int t = tid; t < DK * 32; t += 256) {
        int k = t >> 5;
        int n = t & 31;
        int gn = n0 + n;
        float v = (gn < NC) ? W[(size_t)k * NC + gn] : 0.f;
        __bf16 h = (__bf16)v;
        __bf16 l = (__bf16)(v - (float)h);
        th[n][k] = h;
        tl[n][k] = l;
    }
    __syncthreads();

    // write phase: contiguous k per output row, uint4 (8 bf16) stores
    for (int t = tid; t < 32 * (DK / 8); t += 256) {
        int n = t / (DK / 8);
        int u = t % (DK / 8);
        uint4 vh = *(const uint4*)(&th[n][8 * u]);
        uint4 vl = *(const uint4*)(&tl[n][8 * u]);
        *(uint4*)(WhiT + (size_t)(n0 + n) * DK + 8 * u) = vh;
        *(uint4*)(WloT + (size_t)(n0 + n) * DK + 8 * u) = vl;
    }
}

// ---------------------------------------------------------------------------
// Kernel 1: bf16x3 WMMA GEMM (64x128 tile) + bias -> logits + per-tile LSE stats.
// D = xhi*Whi + xhi*Wlo + xlo*Whi, accumulated in f32 (~f32 accuracy).
// ---------------------------------------------------------------------------
__global__ __launch_bounds__(256)
void gemm_lse_stats(const __bf16* __restrict__ xhi, const __bf16* __restrict__ xlo,
                    const __bf16* __restrict__ WhiT, const __bf16* __restrict__ WloT,
                    const float* __restrict__ bias, float* __restrict__ out,
                    float* __restrict__ pm, float* __restrict__ ps)
{
    __shared__ __align__(16) char smem[2 * TM * APITCH * 2 + 2 * TN * APITCH * 2]; // 55296 B
    __bf16* sAhi = (__bf16*)smem;                    // [TM][APITCH]
    __bf16* sAlo = sAhi + TM * APITCH;
    __bf16* sBhi = sAlo + TM * APITCH;               // [TN][APITCH]
    __bf16* sBlo = sBhi + TN * APITCH;
    float*  Llds = (float*)smem;                     // [TM][TN] reused after K loop
    float*  red  = Llds + TM * TN;                   // [64][4]
    float*  rmax = red + 64 * 4;                     // [64]

    const int tid  = threadIdx.x;
    const int lane = tid & 31;
    const int wave = tid >> 5;
    const int wm   = wave >> 1;          // 0..3 -> 16-row strip
    const int wn   = wave & 1;           // 0..1 -> 64-col strip
    const int hi   = lane >> 4;
    const int lo   = lane & 15;
    const int m0   = blockIdx.y * TM;
    const int n0   = blockIdx.x * TN;

    v8f acc[4] = {};

    for (int kc = 0; kc < DK; kc += TKC) {
        // ---- stage A (x) hi/lo: 64 x 64 bf16, uint4 copies ----
        #pragma unroll
        for (int t = tid; t < TM * (TKC / 8); t += 256) {
            int r = t / (TKC / 8);
            int u = t % (TKC / 8);
            uint4 vh = *(const uint4*)(xhi + (size_t)(m0 + r) * DK + kc + 8 * u);
            uint4 vl = *(const uint4*)(xlo + (size_t)(m0 + r) * DK + kc + 8 * u);
            *(uint4*)(sAhi + r * APITCH + 8 * u) = vh;
            *(uint4*)(sAlo + r * APITCH + 8 * u) = vl;
        }
        // ---- stage B (W^T) hi/lo: 128 x 64 bf16, no bounds checks (padded) ----
        #pragma unroll
        for (int t = tid; t < TN * (TKC / 8); t += 256) {
            int n = t / (TKC / 8);
            int u = t % (TKC / 8);
            uint4 vh = *(const uint4*)(WhiT + (size_t)(n0 + n) * DK + kc + 8 * u);
            uint4 vl = *(const uint4*)(WloT + (size_t)(n0 + n) * DK + kc + 8 * u);
            *(uint4*)(sBhi + n * APITCH + 8 * u) = vh;
            *(uint4*)(sBlo + n * APITCH + 8 * u) = vl;
        }
        __syncthreads();

        // ---- WMMA inner loop: K in steps of 32, 3 bf16 WMMAs per subtile ----
        #pragma unroll
        for (int k = 0; k < TKC; k += 32) {
            // A 16x32 bf16 layout: lanes 0-15 = rows; VGPR0-3 = K[8hi..8hi+7],
            // VGPR4-7 = K[16+8hi..23+8hi]
            const __bf16* arh = sAhi + (wm * 16 + lo) * APITCH + k + 8 * hi;
            const __bf16* arl = sAlo + (wm * 16 + lo) * APITCH + k + 8 * hi;
            v8bf ah0 = *(const v8bf*)(arh);
            v8bf ah1 = *(const v8bf*)(arh + 16);
            v8bf al0 = *(const v8bf*)(arl);
            v8bf al1 = *(const v8bf*)(arl + 16);
            v16bf ahi_f = __builtin_shufflevector(ah0, ah1, 0,1,2,3,4,5,6,7,8,9,10,11,12,13,14,15);
            v16bf alo_f = __builtin_shufflevector(al0, al1, 0,1,2,3,4,5,6,7,8,9,10,11,12,13,14,15);
            #pragma unroll
            for (int s = 0; s < 4; ++s) {
                // B 32x16 bf16 layout: lane%16 = column; per-lane K[8hi..8hi+7], K[16+8hi..23+8hi]
                const __bf16* brh = sBhi + (wn * 64 + s * 16 + lo) * APITCH + k + 8 * hi;
                const __bf16* brl = sBlo + (wn * 64 + s * 16 + lo) * APITCH + k + 8 * hi;
                v8bf bh0 = *(const v8bf*)(brh);
                v8bf bh1 = *(const v8bf*)(brh + 16);
                v8bf bl0 = *(const v8bf*)(brl);
                v8bf bl1 = *(const v8bf*)(brl + 16);
                v16bf bhi_f = __builtin_shufflevector(bh0, bh1, 0,1,2,3,4,5,6,7,8,9,10,11,12,13,14,15);
                v16bf blo_f = __builtin_shufflevector(bl0, bl1, 0,1,2,3,4,5,6,7,8,9,10,11,12,13,14,15);
                acc[s] = __builtin_amdgcn_wmma_f32_16x16x32_bf16(
                    false, ahi_f, false, bhi_f, (short)0, acc[s], false, false);
                acc[s] = __builtin_amdgcn_wmma_f32_16x16x32_bf16(
                    false, ahi_f, false, blo_f, (short)0, acc[s], false, false);
                acc[s] = __builtin_amdgcn_wmma_f32_16x16x32_bf16(
                    false, alo_f, false, bhi_f, (short)0, acc[s], false, false);
            }
        }
        __syncthreads();
    }

    // ---- spill accumulators (+bias) into LDS logits tile ----
    #pragma unroll
    for (int s = 0; s < 4; ++s) {
        int colT = wn * 64 + s * 16 + lo;
        int col  = n0 + colT;
        float bv = (col < NC) ? bias[col] : 0.f;
        #pragma unroll
        for (int i = 0; i < 8; ++i) {
            Llds[(wm * 16 + i + 8 * hi) * TN + colT] = acc[s][i] + bv;
        }
    }
    __syncthreads();

    // ---- per-row tile max (4 threads / row, 32 cols each) ----
    const int row = tid >> 2;
    const int sub = tid & 3;
    {
        float mx = -3.4028235e38f;
        #pragma unroll 4
        for (int c = sub * 32; c < sub * 32 + 32; ++c)
            if (n0 + c < NC) mx = fmaxf(mx, Llds[row * TN + c]);
        red[row * 4 + sub] = mx;
    }
    __syncthreads();
    if (sub == 0) {
        rmax[row] = fmaxf(fmaxf(red[row * 4 + 0], red[row * 4 + 1]),
                          fmaxf(red[row * 4 + 2], red[row * 4 + 3]));
    }
    __syncthreads();

    // ---- per-row tile sum(exp(v - max)) ----
    {
        float m = rmax[row];
        float sacc = 0.f;
        #pragma unroll 4
        for (int c = sub * 32; c < sub * 32 + 32; ++c)
            if (n0 + c < NC) sacc += __expf(Llds[row * TN + c] - m);
        red[row * 4 + sub] = sacc;
    }
    __syncthreads();
    if (sub == 0) {
        float s = red[row * 4 + 0] + red[row * 4 + 1] +
                  red[row * 4 + 2] + red[row * 4 + 3];
        pm[(size_t)(m0 + row) * NTILES + blockIdx.x] = rmax[row];
        ps[(size_t)(m0 + row) * NTILES + blockIdx.x] = s;
    }

    // ---- coalesced logits store (quads never straddle NC; NC % 4 == 0) ----
    #pragma unroll
    for (int t = tid; t < TM * (TN / 4); t += 256) {
        int r = t / (TN / 4);
        int q = t % (TN / 4);
        int col = n0 + 4 * q;
        if (col < NC) {
            const float* src = Llds + r * TN + 4 * q;
            f4 v = {src[0], src[1], src[2], src[3]};
            *(f4*)(out + (size_t)(m0 + r) * NC + col) = v;
        }
    }
}

// ---------------------------------------------------------------------------
// Kernel 2: combine per-tile (m, s) pairs per row -> global (M, S)
// ---------------------------------------------------------------------------
__global__ __launch_bounds__(256)
void reduce_stats(const float* __restrict__ pm, const float* __restrict__ ps,
                  float* __restrict__ rowM, float* __restrict__ rowS)
{
    int row = blockIdx.x * 256 + threadIdx.x;
    if (row >= BATCH) return;
    const float* pmr = pm + (size_t)row * NTILES;
    const float* psr = ps + (size_t)row * NTILES;
    float M = -3.4028235e38f;
    for (int t = 0; t < NTILES; ++t) M = fmaxf(M, pmr[t]);
    float S = 0.f;
    for (int t = 0; t < NTILES; ++t) S += psr[t] * __expf(pmr[t] - M);
    rowM[row] = M;
    rowS[row] = S;
}

// ---------------------------------------------------------------------------
// Kernel 3: c = -(M + log(S - exp(v - M))) in place over the logits, float4.
// ---------------------------------------------------------------------------
__global__ __launch_bounds__(256)
void finalize_lse(float* __restrict__ out, const float* __restrict__ rowM,
                  const float* __restrict__ rowS)
{
    const size_t nq = (size_t)BATCH * NC / 4;
    for (size_t q = (size_t)blockIdx.x * blockDim.x + threadIdx.x; q < nq;
         q += (size_t)gridDim.x * blockDim.x) {
        size_t e = q * 4;
        int row = (int)(e / NC);          // NC % 4 == 0 -> whole quad in one row
        float M = rowM[row];
        float S = rowS[row];
        f4 v = ((const f4*)out)[q];
        f4 r;
        r.x = -(M + __logf(S - __expf(v.x - M)));
        r.y = -(M + __logf(S - __expf(v.y - M)));
        r.z = -(M + __logf(S - __expf(v.z - M)));
        r.w = -(M + __logf(S - __expf(v.w - M)));
        ((f4*)out)[q] = r;
    }
}

// ---------------------------------------------------------------------------
extern "C" void kernel_launch(void* const* d_in, const int* in_sizes, int n_in,
                              void* d_out, int out_size, void* d_ws, size_t ws_size,
                              hipStream_t stream)
{
    const float* x    = (const float*)d_in[0];
    const float* W    = (const float*)d_in[1];
    const float* bias = (const float*)d_in[2];
    float* out = (float*)d_out;

    // workspace layout (16B aligned partitions)
    __bf16* WhiT = (__bf16*)d_ws;                        // [WT_ROWS][DK]
    __bf16* WloT = WhiT + (size_t)WT_ROWS * DK;          // [WT_ROWS][DK]
    __bf16* xhi  = WloT + (size_t)WT_ROWS * DK;          // [BATCH][DK]
    __bf16* xlo  = xhi + (size_t)BATCH * DK;             // [BATCH][DK]
    float*  pm   = (float*)(xlo + (size_t)BATCH * DK);   // [BATCH][NTILES]
    float*  ps   = pm + (size_t)BATCH * NTILES;          // [BATCH][NTILES]
    float*  rowM = ps + (size_t)BATCH * NTILES;          // [BATCH]
    float*  rowS = rowM + BATCH;                         // [BATCH]

    split_x_kernel<<<512, 256, 0, stream>>>(x, xhi, xlo);
    transpose_split_w<<<WT_ROWS / 32, 256, 0, stream>>>(W, WhiT, WloT);

    dim3 g1(NTILES, BATCH / TM);                         // 79 x 128 tiles
    gemm_lse_stats<<<g1, 256, 0, stream>>>(xhi, xlo, WhiT, WloT, bias, out, pm, ps);
    reduce_stats<<<BATCH / 256, 256, 0, stream>>>(pm, ps, rowM, rowS);
    finalize_lse<<<4096, 256, 0, stream>>>(out, rowM, rowS);
}
